// MoEF_17222818857205
// MI455X (gfx1250) — matrix operand order repaired and checked by reading, not compiled, not run on hardware
//
#include <hip/hip_runtime.h>
#include <hip/hip_bf16.h>
#include <math.h>

// Shapes
#define NB 16
#define NC 128
#define NO 128
#define NH 128
#define NW 128
#define NE 3
#define HWSZ (NH * NW)

typedef __attribute__((ext_vector_type(16))) _Float16 v16h;
typedef __attribute__((ext_vector_type(8)))  _Float16 v8h;
typedef __attribute__((ext_vector_type(8)))  float    v8f;

#define RS 72  // LDS row stride in halves (64 data + 8 pad; 144B = 9*16 -> b128 aligned)

static __device__ __forceinline__ v16h cat16(v8h lo, v8h hi) {
    return __builtin_shufflevector(lo, hi, 0, 1, 2, 3, 4, 5, 6, 7,
                                   8, 9, 10, 11, 12, 13, 14, 15);
}

// ---------------- Kernel 1: ctx[b,c] = mean_{h,w} y[b,c,h,w] ----------------
__global__ void moe_ctx_kernel(const float* __restrict__ y, float* __restrict__ ctx) {
    const int bc = blockIdx.x;                 // 0 .. B*C-1
    const float* p = y + (size_t)bc * HWSZ;
    float s = 0.f;
    for (int i = threadIdx.x; i < HWSZ; i += 256) s += p[i];
    __shared__ float red[256];
    red[threadIdx.x] = s;
    __syncthreads();
    for (int off = 128; off > 0; off >>= 1) {
        if (threadIdx.x < off) red[threadIdx.x] += red[threadIdx.x + off];
        __syncthreads();
    }
    if (threadIdx.x == 0) ctx[bc] = red[0] * (1.0f / (float)HWSZ);
}

// ---------------- Kernel 2: gates[b,e] = softmax(ctx @ folded gate_w + b) ----
__global__ void moe_gate_kernel(const float* __restrict__ ctx,
                                const float* __restrict__ gw,
                                const float* __restrict__ gb,
                                float* __restrict__ gates) {
    const int b = blockIdx.x;
    const int c = threadIdx.x;                 // 0..127
    const float v = ctx[b * NC + c];
    __shared__ float red[NE][NC];
    for (int e = 0; e < NE; ++e)
        red[e][c] = v * (gw[c * NE + e] + gw[(c + NC) * NE + e]); // fold dup channels
    __syncthreads();
    for (int off = 64; off > 0; off >>= 1) {
        if (c < off)
            for (int e = 0; e < NE; ++e) red[e][c] += red[e][c + off];
        __syncthreads();
    }
    if (c == 0) {
        float l0 = red[0][0] + gb[0];
        float l1 = red[1][0] + gb[1];
        float l2 = red[2][0] + gb[2];
        float m = fmaxf(l0, fmaxf(l1, l2));
        float e0 = expf(l0 - m), e1 = expf(l1 - m), e2 = expf(l2 - m);
        float inv = 1.0f / (e0 + e1 + e2);
        gates[b * 4 + 0] = e0 * inv;
        gates[b * 4 + 1] = e1 * inv;
        gates[b * 4 + 2] = e2 * inv;
    }
}

// ------- Kernel 3: Wf[b][khkw][o][c] = f16( sum_e g[b,e]*(E[e,o,c]+E[e,o,c+C]) )
__global__ void moe_wagg_kernel(const float* __restrict__ experts,
                                const float* __restrict__ gates,
                                _Float16* __restrict__ Wf) {
    const int bo = blockIdx.x;
    const int b = bo >> 7;
    const int o = bo & 127;
    const int c = threadIdx.x;                 // 0..127
    float g[NE];
    g[0] = gates[b * 4 + 0];
    g[1] = gates[b * 4 + 1];
    g[2] = gates[b * 4 + 2];
    float acc[9];
#pragma unroll
    for (int k = 0; k < 9; ++k) acc[k] = 0.f;
#pragma unroll
    for (int e = 0; e < NE; ++e) {
        const float* p0 = experts + ((size_t)(e * NO + o) * (2 * NC) + c) * 9;
        const float* p1 = p0 + (size_t)NC * 9;  // folded duplicate input half
#pragma unroll
        for (int k = 0; k < 9; ++k) acc[k] += g[e] * (p0[k] + p1[k]);
    }
#pragma unroll
    for (int k = 0; k < 9; ++k)
        Wf[((size_t)(b * 9 + k) * NO + o) * NC + c] = (_Float16)acc[k];
}

// ---------------- Kernel 4: implicit-GEMM conv with WMMA ----------------
// One workgroup per (b, h): M=O=128, N=W=128, K=9*128 in 18 chunks of 64.
// Double-buffered LDS, async A staging, register-prefetched B staging.
__global__ __launch_bounds__(256) void moe_conv_kernel(const float* __restrict__ q,
                                                       const _Float16* __restrict__ Wf,
                                                       float* __restrict__ out) {
    __shared__ _Float16 As[2][NO * RS];   // [buf][o][k_in_chunk]
    __shared__ _Float16 Bs[2][NW * RS];   // [buf][n][k_in_chunk]

    const int bid = blockIdx.x;
    const int b = bid >> 7;
    const int h = bid & 127;
    const int t = threadIdx.x;
    const int lane = t & 31;
    const int wave = t >> 5;
    const int hi = lane >> 4;
    const int ml = lane & 15;
    const int m0 = wave * 16;

    // B-staging thread mapping (fixed across chunks)
    const int p  = t & 31;             // channel-pair index within chunk
    const int nb = (t >> 5) << 4;      // 16 pixels per thread
    // A-staging thread mapping
    const int ao   = t >> 1;           // o row
    const int aksb = (t & 1) * 32;     // k offset within chunk (32 halves = 64B)

    v8f acc[8];
#pragma unroll
    for (int i = 0; i < 8; ++i) acc[i] = (v8f){0.f, 0.f, 0.f, 0.f, 0.f, 0.f, 0.f, 0.f};

    float f0[16], f1[16];

    auto stageA_async = [&](int buf, int khkw, int c0) {
        const _Float16* src =
            Wf + ((size_t)(b * 9 + khkw) * NO + ao) * NC + c0 + aksb;
        unsigned long long gaddr = (unsigned long long)src;
        unsigned lds = (unsigned)(size_t)(&As[buf][ao * RS + aksb]);
        asm volatile(
            "global_load_async_to_lds_b128 %0, %1, off\n\t"
            "global_load_async_to_lds_b128 %0, %1, off offset:16\n\t"
            "global_load_async_to_lds_b128 %0, %1, off offset:32\n\t"
            "global_load_async_to_lds_b128 %0, %1, off offset:48"
            :: "v"(lds), "v"(gaddr) : "memory");
    };

    auto loadB_regs = [&](int kw, int hq, bool hok, int c0) {
        const int cA = c0 + 2 * p;
        const float* q0 = q + ((size_t)(b * NC + cA) * NH + (hok ? hq : 0)) * NW;
        const float* q1 = q0 + HWSZ;
#pragma unroll
        for (int j = 0; j < 16; ++j) {
            const int w = nb + j + kw - 1;
            const bool ok = hok && (w >= 0) && (w < NW);
            f0[j] = ok ? q0[w] : 0.f;
            f1[j] = ok ? q1[w] : 0.f;
        }
    };

    // ---- prologue: stage chunk 0 ----
    stageA_async(0, /*khkw=*/0, /*c0=*/0);
    {
        const int hq0 = h - 1;
        loadB_regs(/*kw=*/0, hq0, (hq0 >= 0), /*c0=*/0);
    }

    for (int chunk = 0; chunk < 18; ++chunk) {
        const int cur = chunk & 1;

        // ---- commit prefetched B registers to LDS (f32->f16 packed) ----
#pragma unroll
        for (int j = 0; j < 16; ++j) {
            union { _Float16 hh[2]; unsigned u; } pk;
            pk.hh[0] = (_Float16)f0[j];
            pk.hh[1] = (_Float16)f1[j];
            *(unsigned*)(&Bs[cur][(nb + j) * RS + 2 * p]) = pk.u;
        }

        // async A tile for this chunk must have landed before the barrier
        asm volatile("s_wait_asynccnt 0x0" ::: "memory");
        __syncthreads();

        // ---- prefetch chunk+1 into the other buffer / registers ----
        if (chunk < 17) {
            const int nchunk = chunk + 1;
            const int khkw = nchunk >> 1;
            const int c0 = (nchunk & 1) * 64;
            const int kh = khkw / 3;
            const int kw = khkw - kh * 3;
            const int hq = h + kh - 1;
            stageA_async(1 - cur, khkw, c0);
            loadB_regs(kw, hq, (hq >= 0) && (hq < NH), c0);
        }

        // ---- compute on buf[cur]: 2 K-steps of 32, 8 N-tiles ----
        const _Float16* Ab = &As[cur][0];
        const _Float16* Bb = &Bs[cur][0];
#pragma unroll
        for (int ks = 0; ks < 64; ks += 32) {
            // A fragment (ISA 16-bit A 16x32 layout: K = hi*8 + {0..7, 16..23})
            const _Float16* arow = Ab + (m0 + ml) * RS + ks + hi * 8;
            v16h a = cat16(*(const v8h*)(arow), *(const v8h*)(arow + 16));
            // all 8 B fragments into distinct registers
            v16h bf[8];
#pragma unroll
            for (int nt = 0; nt < 8; ++nt) {
                const _Float16* brow = Bb + (nt * 16 + ml) * RS + ks + hi * 8;
                bf[nt] = cat16(*(const v8h*)(brow), *(const v8h*)(brow + 16));
            }
            // Fence: force all 18 ds_load_b128 to batch + complete here so the
            // scheduler cannot re-serialize into load->wait->wmma pairs; the 8
            // WMMAs below then issue back-to-back after a single dscnt wait.
            asm(""
                : "+v"(a), "+v"(bf[0]), "+v"(bf[1]), "+v"(bf[2]), "+v"(bf[3]),
                  "+v"(bf[4]), "+v"(bf[5]), "+v"(bf[6]), "+v"(bf[7]));
#pragma unroll
            for (int nt = 0; nt < 8; ++nt)
                acc[nt] = __builtin_amdgcn_wmma_f32_16x16x32_f16(
                    false, a, false, bf[nt], (short)0, acc[nt], false, false);
        }

        __syncthreads();
    }

    // ---- store: C/D layout M = v + 8*hi, N = lane&15 ----
#pragma unroll
    for (int nt = 0; nt < 8; ++nt) {
        const int w = nt * 16 + ml;
#pragma unroll
        for (int v = 0; v < 8; ++v) {
            const int o = m0 + hi * 8 + v;
            out[((size_t)(b * NO + o) * NH + h) * NW + w] = acc[nt][v];
        }
    }
}

extern "C" void kernel_launch(void* const* d_in, const int* in_sizes, int n_in,
                              void* d_out, int out_size, void* d_ws, size_t ws_size,
                              hipStream_t stream) {
    (void)in_sizes; (void)n_in; (void)out_size; (void)ws_size;
    const float* q       = (const float*)d_in[0];
    const float* y       = (const float*)d_in[1];
    const float* experts = (const float*)d_in[2];
    const float* gate_w  = (const float*)d_in[3];
    const float* gate_b  = (const float*)d_in[4];
    float* out = (float*)d_out;

    // workspace layout
    float*     ctx   = (float*)d_ws;                          // B*C f32 (8 KB)
    float*     gates = ctx + NB * NC;                         // B*4 f32
    _Float16*  Wf    = (_Float16*)((char*)d_ws + 16384);      // B*9*O*C f16 (4.7 MB)

    moe_ctx_kernel <<<NB * NC, 256, 0, stream>>>(y, ctx);
    moe_gate_kernel<<<NB, NC, 0, stream>>>(ctx, gate_w, gate_b, gates);
    moe_wagg_kernel<<<NB * NO, NC, 0, stream>>>(experts, gates, Wf);
    moe_conv_kernel<<<NB * NH, 256, 0, stream>>>(q, Wf, out);
}
